// DownsampleLayer_27127013441594
// MI455X (gfx1250) — matrix operand order repaired and checked by reading, not compiled, not run on hardware
//
#include <hip/hip_runtime.h>
#include <hip/hip_bf16.h>

// Problem constants (fixed by the reference)
#define BN    4
#define NPTS  8192
#define MPTS  2048
#define CDIM  64
#define KNN   16
#define NG    8
#define NEGF  (-3.402823466e38f)

typedef __attribute__((ext_vector_type(16))) _Float16 v16h;
typedef __attribute__((ext_vector_type(8)))  float    v8f;
typedef unsigned int u32x4 __attribute__((ext_vector_type(4)));
typedef int          i32x4 __attribute__((ext_vector_type(4)));
typedef int          i32x8 __attribute__((ext_vector_type(8)));

// ---------------------------------------------------------------- utilities
__device__ __forceinline__ float wave_sum(float v) {
    #pragma unroll
    for (int o = 16; o > 0; o >>= 1) v += __shfl_down(v, o);
    return v;
}

// ---------------------------------------------------------------- FPS
// One workgroup per batch. Both the running distance array (32 KB) and the
// whole point cloud (96 KB) live in dynamic LDS -- 128 KB of the 320 KB/WGP.
extern __shared__ float fps_smem[];
__global__ __launch_bounds__(1024) void fps_kernel(const float* __restrict__ xyz,
                                                   int* __restrict__ sidx) {
    float* dist = fps_smem;          // NPTS
    float* pts  = fps_smem + NPTS;   // 3*NPTS
    __shared__ float wmax[32];
    __shared__ int   warg[32];
    __shared__ int   scur;
    const int b = blockIdx.x, tid = threadIdx.x;
    const float* xb = xyz + (size_t)b * 3 * NPTS;
    for (int i = tid; i < 3 * NPTS; i += 1024) pts[i] = xb[i];
    for (int i = tid; i < NPTS; i += 1024) dist[i] = 1e10f;
    if (tid == 0) { scur = 0; sidx[b * MPTS] = 0; }
    __syncthreads();
    for (int m = 1; m < MPTS; ++m) {
        const int last = scur;
        const float lx = pts[last], ly = pts[NPTS + last], lz = pts[2 * NPTS + last];
        float bd = -1.f; int bi = 0;
        for (int i = tid; i < NPTS; i += 1024) {
            float dx = pts[i] - lx, dy = pts[NPTS + i] - ly, dz = pts[2 * NPTS + i] - lz;
            float d = dx * dx + dy * dy + dz * dz;
            float dd = fminf(dist[i], d);
            dist[i] = dd;
            if (dd > bd) { bd = dd; bi = i; }
        }
        #pragma unroll
        for (int o = 16; o > 0; o >>= 1) {
            float od = __shfl_down(bd, o); int oi = __shfl_down(bi, o);
            if (od > bd || (od == bd && oi < bi)) { bd = od; bi = oi; }
        }
        const int w = tid >> 5;
        if ((tid & 31) == 0) { wmax[w] = bd; warg[w] = bi; }
        __syncthreads();
        if (tid < 32) {
            bd = wmax[tid]; bi = warg[tid];
            #pragma unroll
            for (int o = 16; o > 0; o >>= 1) {
                float od = __shfl_down(bd, o); int oi = __shfl_down(bi, o);
                if (od > bd || (od == bd && oi < bi)) { bd = od; bi = oi; }
            }
            if (tid == 0) { scur = bi; sidx[b * MPTS + m] = bi; }
        }
        __syncthreads();
    }
}

// ------------------------------------------------- gathers / elementwise
__global__ void gather_kernel(float* __restrict__ out, const float* __restrict__ src,
                              const int* __restrict__ sidx, int C, int srcP) {
    size_t idx = (size_t)blockIdx.x * blockDim.x + threadIdx.x;
    size_t total = (size_t)BN * C * MPTS;
    if (idx >= total) return;
    int m = (int)(idx % MPTS); size_t t = idx / MPTS;
    int c = (int)(t % C); int b = (int)(t / C);
    out[idx] = src[((size_t)b * C + c) * srcP + sidx[b * MPTS + m]];
}

__global__ void addgather_kernel(float* __restrict__ out, const float* __restrict__ src,
                                 const int* __restrict__ sidx) {
    size_t idx = (size_t)blockIdx.x * blockDim.x + threadIdx.x;
    size_t total = (size_t)BN * CDIM * MPTS;
    if (idx >= total) return;
    int m = (int)(idx % MPTS); size_t t = idx / MPTS;
    int c = (int)(t % CDIM); int b = (int)(t / CDIM);
    out[idx] += src[((size_t)b * CDIM + c) * NPTS + sidx[b * MPTS + m]];
}

__global__ void add_kernel(float* __restrict__ out, const float* __restrict__ a, long long total) {
    size_t idx = (size_t)blockIdx.x * blockDim.x + threadIdx.x;
    if (idx >= (size_t)total) return;
    out[idx] += a[idx];
}

// ------------------------------------------------- nearest centroid / kNN
__global__ __launch_bounds__(256) void nn_kernel(const float* __restrict__ xyz,
                                                 const float* __restrict__ sxyz,
                                                 int* __restrict__ nn) {
    __shared__ float sx[256], sy[256], sz[256];
    int g = blockIdx.x * 256 + threadIdx.x;
    int b = g / NPTS, n = g % NPTS;
    const float* xb = xyz + (size_t)b * 3 * NPTS;
    const float* sb = sxyz + (size_t)b * 3 * MPTS;
    float px = xb[n], py = xb[NPTS + n], pz = xb[2 * NPTS + n];
    float bd = 3.4e38f; int bi = 0;
    for (int m0 = 0; m0 < MPTS; m0 += 256) {
        __syncthreads();
        sx[threadIdx.x] = sb[m0 + threadIdx.x];
        sy[threadIdx.x] = sb[MPTS + m0 + threadIdx.x];
        sz[threadIdx.x] = sb[2 * MPTS + m0 + threadIdx.x];
        __syncthreads();
        if (m0 + 256 < MPTS) __builtin_prefetch(sb + m0 + 256, 0, 0);
        for (int j = 0; j < 256; ++j) {
            float dx = px - sx[j], dy = py - sy[j], dz = pz - sz[j];
            float d = dx * dx + dy * dy + dz * dz;
            if (d < bd) { bd = d; bi = m0 + j; }
        }
    }
    nn[g] = bi;
}

__global__ __launch_bounds__(256) void knn_kernel(const float* __restrict__ xyz,
                                                  const float* __restrict__ sxyz,
                                                  int* __restrict__ knn) {
    __shared__ float sx[256], sy[256], sz[256];
    int g = blockIdx.x * 256 + threadIdx.x;
    int b = g / MPTS, m = g % MPTS;
    const float* sb = sxyz + (size_t)b * 3 * MPTS;
    const float* xb = xyz + (size_t)b * 3 * NPTS;
    float cx = sb[m], cy = sb[MPTS + m], cz = sb[2 * MPTS + m];
    float bd[KNN]; int bi[KNN];
    #pragma unroll
    for (int k = 0; k < KNN; ++k) { bd[k] = 3.4e38f; bi[k] = 0; }
    for (int n0 = 0; n0 < NPTS; n0 += 256) {
        __syncthreads();
        sx[threadIdx.x] = xb[n0 + threadIdx.x];
        sy[threadIdx.x] = xb[NPTS + n0 + threadIdx.x];
        sz[threadIdx.x] = xb[2 * NPTS + n0 + threadIdx.x];
        __syncthreads();
        if (n0 + 256 < NPTS) __builtin_prefetch(xb + n0 + 256, 0, 0);
        for (int j = 0; j < 256; ++j) {
            float dx = cx - sx[j], dy = cy - sy[j], dz = cz - sz[j];
            float d = dx * dx + dy * dy + dz * dz;
            if (d < bd[KNN - 1]) {
                int pos = KNN - 1;
                while (pos > 0 && bd[pos - 1] > d) {
                    bd[pos] = bd[pos - 1]; bi[pos] = bi[pos - 1]; --pos;
                }
                bd[pos] = d; bi[pos] = n0 + j;
            }
        }
    }
    #pragma unroll
    for (int k = 0; k < KNN; ++k) knn[(size_t)g * KNN + k] = bi[k];
}

__global__ void maskdist_kernel(const float* __restrict__ xyz, const float* __restrict__ sxyz,
                                const int* __restrict__ knn, const int* __restrict__ nn,
                                float* __restrict__ mask, float* __restrict__ dnum_out,
                                float* __restrict__ mdist_out) {
    int g = blockIdx.x * blockDim.x + threadIdx.x;
    if (g >= BN * MPTS) return;
    int b = g / MPTS, m = g % MPTS;
    const float* xb = xyz + (size_t)b * 3 * NPTS;
    const float* sb = sxyz + (size_t)b * 3 * MPTS;
    float cx = sb[m], cy = sb[MPTS + m], cz = sb[2 * MPTS + m];
    float dsum = 0.f, cnt = 0.f;
    #pragma unroll
    for (int k = 0; k < KNN; ++k) {
        int j = knn[(size_t)g * KNN + k];
        float mk = (nn[b * NPTS + j] == m) ? 1.f : 0.f;
        mask[(size_t)g * KNN + k] = mk;
        float dx = xb[j] - cx, dy = xb[NPTS + j] - cy, dz = xb[2 * NPTS + j] - cz;
        float d2 = dx * dx + dy * dy + dz * dz;
        float d = d2 > 0.f ? sqrtf(d2) : 0.f;
        dsum += d * mk; cnt += mk;
    }
    dnum_out[g] = cnt;
    mdist_out[g] = dsum / cnt;
}

// ------------------------------------------------- WMMA conv1x1 (GEMM)
// out(b,o,p) = sum_i W[o,i] * in(b,i,p) + bias[o], O = 64 always, I compile-time.
// Weights are DMA'd to LDS with the Tensor Data Mover, then swizzled once into
// A-fragment order (f16).  Plain/concat activations are staged as f16 tiles in
// LDS [point][k] so B fragments are contiguous ds loads.
struct GemmP {
    const float* W; const float* bias; float* out;
    const float* in; const float* in2; const float* in3;
    const float* sxyz; const float* xyz; const int* knn;
    const float* query; const float* keyfull; const float* posenc;
    int P;
};

template <int MODE, int I>
__global__ __launch_bounds__(256) void gemm_wmma_kernel(GemmP p) {
    constexpr int IP = (I + 31) & ~31;
    __shared__ float    Wraw[64 * I];      // raw f32 weights (TDM destination)
    __shared__ _Float16 Wswz[64 * IP];     // A-fragment-ordered f16 weights
    __shared__ _Float16 Bt[32 * 32];       // activation tile [point][k]
    const int tid = threadIdx.x;
    const int lane = tid & 31, wid = tid >> 5;
    const int b = blockIdx.y;
    const int pbase = blockIdx.x * 32;
    const int g = lane >> 4, r = lane & 15;

    // ---- preload this wave's 8 bias values (batched, overlaps staging) ----
    float brow[8];
    #pragma unroll
    for (int jj = 0; jj < 8; ++jj) brow[jj] = p.bias[(wid & 3) * 16 + jj + 8 * g];

    // ---- TDM: async DMA of the 64 x I f32 weight matrix into LDS (wave 0) ----
    if (wid == 0) {
        unsigned long long ga = (unsigned long long)(size_t)p.W;
        unsigned ldsoff = (unsigned)(size_t)&Wraw[0];
        // D# group0: count=1 | lds_addr | global_addr | type=2
        u32x4 g0 = { 1u, ldsoff, (unsigned)(ga & 0xFFFFFFFFull),
                     (unsigned)((ga >> 32) & 0x01FFFFFFull) | 0x80000000u };
        // D# group1: data_size=4B, tensor I x 64, tile I x 64, dim0 stride = I
        i32x8 g1 = { (int)0x00020000, (int)(I << 16), (int)(64 << 16), (int)(I << 16),
                     64, I, 0, 0 };
        i32x4 z4 = { 0, 0, 0, 0 };
        i32x8 z8 = { 0, 0, 0, 0, 0, 0, 0, 0 };
        __builtin_amdgcn_tensor_load_to_lds(g0, g1, z4, z4, z8, 0);
        __builtin_amdgcn_s_wait_tensorcnt(0);
    }
    __syncthreads();
    // ---- swizzle weights into per-lane A-fragment order, zero-pad K to IP ----
    for (int t = tid; t < 64 * IP; t += 256) {
        int e = t & 15, ln = (t >> 4) & 31, ot = (t >> 9) & 3, kb = t >> 11;
        int gg = ln >> 4, rr = ln & 15, v = e >> 1, bit = e & 1;
        int ka = 2 * v + bit + 16 * gg;
        int k = kb * 32 + ka, o = ot * 16 + rr;
        Wswz[t] = (_Float16)((k < I) ? Wraw[o * I + k] : 0.f);
    }
    __syncthreads();

    v8f acc = {};
    for (int k0 = 0; k0 < IP; k0 += 32) {
        v16h af, bf;
        if constexpr (MODE == 0 || MODE == 4) {
            __syncthreads();
            for (int t = tid; t < 1024; t += 256) {
                int row = t >> 5, col = t & 31;            // row = k, col = point
                int ch = k0 + row;
                int chc = (ch < I) ? ch : I - 1;           // clamp: load always valid
                float val;
                if constexpr (MODE == 4) {
                    const float* s = (chc < CDIM) ? p.in
                                   : (chc < 2 * CDIM ? p.in2 : p.in3);
                    val = s[((size_t)b * CDIM + (chc & 63)) * p.P + pbase + col];
                } else {
                    val = p.in[((size_t)b * I + chc) * p.P + pbase + col];
                }
                Bt[col * 32 + row] = (_Float16)((ch < I) ? val : 0.f);
            }
            __syncthreads();
            const _Float16* bp = &Bt[((wid >> 2) * 16 + r) * 32 + 16 * g];
            #pragma unroll
            for (int e = 0; e < 16; ++e) bf[e] = bp[e];
        } else {
            const int pp = pbase + (wid >> 2) * 16 + r;
            const int m = pp >> 4, kk = pp & 15;
            const int j = p.knn[((size_t)b * MPTS + m) * KNN + kk];
            if constexpr (MODE == 1) {                      // sxyz - gathered xyz
                float d0 = p.sxyz[((size_t)b * 3 + 0) * MPTS + m] - p.xyz[((size_t)b * 3 + 0) * NPTS + j];
                float d1 = p.sxyz[((size_t)b * 3 + 1) * MPTS + m] - p.xyz[((size_t)b * 3 + 1) * NPTS + j];
                float d2 = p.sxyz[((size_t)b * 3 + 2) * MPTS + m] - p.xyz[((size_t)b * 3 + 2) * NPTS + j];
                #pragma unroll
                for (int e = 0; e < 16; ++e) {
                    int ib = e + 16 * g;
                    float val = (ib == 0) ? d0 : (ib == 1) ? d1 : (ib == 2) ? d2 : 0.f;
                    bf[e] = (_Float16)val;
                }
            } else if constexpr (MODE == 3) {               // direction(3), dist
                float o0 = p.xyz[((size_t)b * 3 + 0) * NPTS + j] - p.sxyz[((size_t)b * 3 + 0) * MPTS + m];
                float o1 = p.xyz[((size_t)b * 3 + 1) * NPTS + j] - p.sxyz[((size_t)b * 3 + 1) * MPTS + m];
                float o2 = p.xyz[((size_t)b * 3 + 2) * NPTS + j] - p.sxyz[((size_t)b * 3 + 2) * MPTS + m];
                float d2 = o0 * o0 + o1 * o1 + o2 * o2;
                float d = d2 > 0.f ? sqrtf(d2) : 0.f;
                float inv = 1.f / fmaxf(d, 1e-12f);
                float l0 = o0 * inv, l1 = o1 * inv, l2 = o2 * inv, l3 = d;
                #pragma unroll
                for (int e = 0; e < 16; ++e) {
                    int ib = e + 16 * g;
                    float val = (ib == 0) ? l0 : (ib == 1) ? l1 : (ib == 2) ? l2
                              : (ib == 3) ? l3 : 0.f;
                    bf[e] = (_Float16)val;
                }
            } else {                                        // MODE 2: q - k + pos_enc
                #pragma unroll
                for (int e = 0; e < 16; ++e) {
                    int i = k0 + e + 16 * g;                // always < 64
                    float val = p.query[((size_t)b * CDIM + i) * MPTS + m]
                              - p.keyfull[((size_t)b * CDIM + i) * NPTS + j]
                              + p.posenc[(((size_t)b * CDIM + i) * MPTS + m) * KNN + kk];
                    bf[e] = (_Float16)val;
                }
            }
        }
        const _Float16* ap = &Wswz[(((k0 >> 5) * 4 + (wid & 3)) * 32 + lane) * 16];
        #pragma unroll
        for (int e = 0; e < 16; ++e) af[e] = ap[e];
        acc = __builtin_amdgcn_wmma_f32_16x16x32_f16(false, af, false, bf,
                                                     (short)0, acc, false, false);
    }
    #pragma unroll
    for (int jj = 0; jj < 8; ++jj) {
        const int o = (wid & 3) * 16 + jj + 8 * g;
        p.out[((size_t)b * CDIM + o) * p.P + (pbase + (wid >> 2) * 16 + r)] =
            acc[jj] + brow[jj];
    }
}

// ------------------------------------------------- GroupNorm
__global__ __launch_bounds__(256) void gn_stats_kernel(const float* __restrict__ x,
                                                       float* __restrict__ stats, long long S) {
    const int bg = blockIdx.x;                         // b*NG + g
    const int cpg = CDIM / NG;                         // 8 channels / group
    const size_t base = (size_t)bg * cpg * S;          // contiguous group slice
    const size_t cnt = (size_t)cpg * S;
    float s = 0.f, ss = 0.f;
    for (size_t t = threadIdx.x; t < cnt; t += 256) {
        float v = x[base + t]; s += v; ss += v * v;
    }
    __shared__ float as[8], ass[8];
    s = wave_sum(s); ss = wave_sum(ss);
    const int w = threadIdx.x >> 5;
    if ((threadIdx.x & 31) == 0) { as[w] = s; ass[w] = ss; }
    __syncthreads();
    if (threadIdx.x == 0) {
        float S1 = 0.f, S2 = 0.f;
        #pragma unroll
        for (int i = 0; i < 8; ++i) { S1 += as[i]; S2 += ass[i]; }
        float mu = S1 / (float)cnt;
        float var = S2 / (float)cnt - mu * mu;
        stats[bg * 2] = mu;
        stats[bg * 2 + 1] = rsqrtf(fmaxf(var, 0.f) + 1e-5f);
    }
}

__global__ void gn_apply_kernel(float* __restrict__ x, const float* __restrict__ stats,
                                const float* __restrict__ gw, const float* __restrict__ gb,
                                long long S) {
    size_t idx = (size_t)blockIdx.x * blockDim.x + threadIdx.x;
    size_t total = (size_t)BN * CDIM * S;
    if (idx >= total) return;
    int c = (int)((idx / S) % CDIM);
    int b = (int)(idx / ((size_t)S * CDIM));
    int grp = c / (CDIM / NG);
    float mu = stats[(b * NG + grp) * 2], rs = stats[(b * NG + grp) * 2 + 1];
    float v = (x[idx] - mu) * rs * gw[c] + gb[c];
    x[idx] = fmaxf(v, 0.f);                            // mlp2 always ReLUs after GN
}

// ------------------------------------------------- softmax over K (masked)
__global__ void softmax_kernel(float* __restrict__ a, const float* __restrict__ mask, float scale) {
    size_t g = (size_t)blockIdx.x * blockDim.x + threadIdx.x;   // (b*C+c)*M + m
    if (g >= (size_t)BN * CDIM * MPTS) return;
    int m = (int)(g % MPTS);
    int b = (int)(g / ((size_t)CDIM * MPTS));
    size_t base = g * KNN;
    size_t mb = ((size_t)b * MPTS + m) * KNN;
    float v[KNN]; float mx = -3.4e38f;
    #pragma unroll
    for (int k = 0; k < KNN; ++k) {
        float val = a[base + k] * scale;
        val = (mask[mb + k] > 0.5f) ? val : NEGF;
        v[k] = val; mx = fmaxf(mx, val);
    }
    float s = 0.f;
    #pragma unroll
    for (int k = 0; k < KNN; ++k) { float e = __expf(v[k] - mx); v[k] = e; s += e; }
    float inv = 1.f / s;
    #pragma unroll
    for (int k = 0; k < KNN; ++k) a[base + k] = v[k] * inv;
}

// ------------------------------------------------- attention-weighted sums
__global__ void wsum_feat_kernel(const float* __restrict__ attn, const float* __restrict__ vfull,
                                 const float* __restrict__ pe, const int* __restrict__ knn,
                                 float* __restrict__ out) {
    size_t g = (size_t)blockIdx.x * blockDim.x + threadIdx.x;
    if (g >= (size_t)BN * CDIM * MPTS) return;
    int m = (int)(g % MPTS); size_t t = g / MPTS;
    int c = (int)(t % CDIM); int b = (int)(t / CDIM);
    const int* kb = knn + ((size_t)b * MPTS + m) * KNN;
    const float* vb = vfull + ((size_t)b * CDIM + c) * NPTS;
    size_t ab = g * KNN;
    float s = 0.f;
    #pragma unroll
    for (int k = 0; k < KNN; ++k) s += attn[ab + k] * (vb[kb[k]] + pe[ab + k]);
    out[g] = s;
}

__global__ void wsum_pe_kernel(const float* __restrict__ pe, const float* __restrict__ attn,
                               float* __restrict__ out) {
    size_t g = (size_t)blockIdx.x * blockDim.x + threadIdx.x;
    if (g >= (size_t)BN * CDIM * MPTS) return;
    size_t ab = g * KNN;
    float s = 0.f;
    #pragma unroll
    for (int k = 0; k < KNN; ++k) s += pe[ab + k] * attn[ab + k];
    out[g] = s;
}

// ---------------------------------------------------------------- driver
extern "C" void kernel_launch(void* const* d_in, const int* in_sizes, int n_in,
                              void* d_out, int out_size, void* d_ws, size_t ws_size,
                              hipStream_t stream) {
    (void)in_sizes; (void)n_in; (void)out_size; (void)ws_size;
    const float* xyz   = (const float*)d_in[0];
    const float* feats = (const float*)d_in[1];
    int ii = 2;
    auto nxt = [&]() { return (const float*)d_in[ii++]; };
    const float *pre_w = nxt(), *pre_b = nxt();
    const float *wq_w = nxt(), *wq_b = nxt();
    const float *wk_w = nxt(), *wk_b = nxt();
    const float *wv_w = nxt(), *wv_b = nxt();
    const float *dl_c1w = nxt(), *dl_c1b = nxt(), *dl_gng = nxt(), *dl_gnb = nxt(),
                *dl_c2w = nxt(), *dl_c2b = nxt();
    const float *gm_c1w = nxt(), *gm_c1b = nxt(), *gm_gng = nxt(), *gm_gnb = nxt(),
                *gm_c2w = nxt(), *gm_c2b = nxt();
    const float *pt_pw = nxt(), *pt_pb = nxt();
    const float *em_c1w = nxt(), *em_c1b = nxt(), *em_gng = nxt(), *em_gnb = nxt(),
                *em_c2w = nxt(), *em_c2b = nxt();
    const float *at_c1w = nxt(), *at_c1b = nxt(), *at_gng = nxt(), *at_gnb = nxt(),
                *at_c2w = nxt(), *at_c2b = nxt();
    const float *dn_c1w = nxt(), *dn_c1b = nxt(), *dn_gng = nxt(), *dn_gnb = nxt(),
                *dn_c2w = nxt(), *dn_c2b = nxt();
    const float *po_w = nxt(), *po_b = nxt();

    // d_out regions in reference tuple order
    float* out = (float*)d_out;
    float* o_sxyz  = out;
    float* o_feat  = o_sxyz + (size_t)BN * 3 * MPTS;
    float* o_dnum  = o_feat + (size_t)BN * CDIM * MPTS;
    float* o_mdist = o_dnum + (size_t)BN * MPTS;

    // workspace carve
    float* ws = (float*)d_ws;
    size_t off = 0;
    auto alloc = [&](size_t n) { float* p = ws + off; off += n; return p; };
    int*   sidx  = (int*)alloc((size_t)BN * MPTS);
    int*   nnid  = (int*)alloc((size_t)BN * NPTS);
    int*   knn   = (int*)alloc((size_t)BN * MPTS * KNN);
    float* maskb = alloc((size_t)BN * MPTS * KNN);
    float* stats = alloc((size_t)BN * NG * 2);
    float* f     = alloc((size_t)BN * CDIM * NPTS);
    float* kful  = alloc((size_t)BN * CDIM * NPTS);
    float* vful  = alloc((size_t)BN * CDIM * NPTS);
    float* sf    = alloc((size_t)BN * CDIM * MPTS);
    float* qry   = alloc((size_t)BN * CDIM * MPTS);
    float* res   = alloc((size_t)BN * CDIM * MPTS);
    float* efeat = alloc((size_t)BN * CDIM * MPTS);
    float* epos  = alloc((size_t)BN * CDIM * MPTS);
    float* edens = alloc((size_t)BN * CDIM * MPTS);
    float* hidm  = alloc((size_t)BN * CDIM * MPTS);
    float* BIG0  = alloc((size_t)BN * CDIM * MPTS * KNN);
    float* BIG1  = alloc((size_t)BN * CDIM * MPTS * KNN);
    float* BIG2  = alloc((size_t)BN * CDIM * MPTS * KNN);

    const long long MK = (long long)MPTS * KNN;
    const long long CM = (long long)BN * CDIM * MPTS;

    auto mk = [&](const float* W, const float* bias, float* outp, int P,
                  const float* in = nullptr, const float* in2 = nullptr,
                  const float* in3 = nullptr, const float* q = nullptr,
                  const float* kf = nullptr, const float* pe = nullptr) {
        GemmP p{};
        p.W = W; p.bias = bias; p.out = outp; p.in = in; p.in2 = in2; p.in3 = in3;
        p.sxyz = o_sxyz; p.xyz = xyz; p.knn = knn;
        p.query = q; p.keyfull = kf; p.posenc = pe;
        p.P = P;
        return p;
    };
    auto do_gn = [&](float* x, const float* gw, const float* gb, long long S) {
        gn_stats_kernel<<<BN * NG, 256, 0, stream>>>(x, stats, S);
        long long tot = (long long)BN * CDIM * S;
        gn_apply_kernel<<<(unsigned)((tot + 255) / 256), 256, 0, stream>>>(x, stats, gw, gb, S);
    };
    const dim3 gN(NPTS / 32, BN), gM(MPTS / 32, BN), gMK((unsigned)(MK / 32), BN);

    // 1. FPS + sampled_xyzs (dynamic LDS: dist + staged cloud = 128 KB)
    fps_kernel<<<BN, 1024, 4u * 4u * NPTS, stream>>>(xyz, sidx);
    gather_kernel<<<(BN * 3 * MPTS + 255) / 256, 256, 0, stream>>>(o_sxyz, xyz, sidx, 3, NPTS);

    // 2. nearest centroid per point, kNN per centroid, mask/dnum/mean-dist
    nn_kernel<<<BN * NPTS / 256, 256, 0, stream>>>(xyz, o_sxyz, nnid);
    knn_kernel<<<BN * MPTS / 256, 256, 0, stream>>>(xyz, o_sxyz, knn);
    maskdist_kernel<<<(BN * MPTS + 255) / 256, 256, 0, stream>>>(xyz, o_sxyz, knn, nnid,
                                                                 maskb, o_dnum, o_mdist);

    // 3. pre-conv, q/k/v projections
    gemm_wmma_kernel<0, 64><<<gN, 256, 0, stream>>>(mk(pre_w, pre_b, f, NPTS, feats));
    gather_kernel<<<(unsigned)((CM + 255) / 256), 256, 0, stream>>>(sf, f, sidx, CDIM, NPTS);
    gemm_wmma_kernel<0, 64><<<gM, 256, 0, stream>>>(mk(wq_w, wq_b, qry, MPTS, sf));
    gemm_wmma_kernel<0, 64><<<gN, 256, 0, stream>>>(mk(wk_w, wk_b, kful, NPTS, f));
    gemm_wmma_kernel<0, 64><<<gN, 256, 0, stream>>>(mk(wv_w, wv_b, vful, NPTS, f));

    // 4. delta MLP (pos_enc): fused diff input
    gemm_wmma_kernel<1, 3><<<gMK, 256, 0, stream>>>(mk(dl_c1w, dl_c1b, BIG0, (int)MK));
    do_gn(BIG0, dl_gng, dl_gnb, MK);
    gemm_wmma_kernel<0, 64><<<gMK, 256, 0, stream>>>(mk(dl_c2w, dl_c2b, BIG1, (int)MK, BIG0));

    // 5. gamma MLP on (q - k + pos_enc), masked softmax, weighted sum
    gemm_wmma_kernel<2, 64><<<gMK, 256, 0, stream>>>(
        mk(gm_c1w, gm_c1b, BIG0, (int)MK, nullptr, nullptr, nullptr, qry, kful, BIG1));
    do_gn(BIG0, gm_gng, gm_gnb, MK);
    gemm_wmma_kernel<0, 64><<<gMK, 256, 0, stream>>>(mk(gm_c2w, gm_c2b, BIG2, (int)MK, BIG0));
    softmax_kernel<<<(unsigned)((CM + 255) / 256), 256, 0, stream>>>(BIG2, maskb, 0.125f);
    wsum_feat_kernel<<<(unsigned)((CM + 255) / 256), 256, 0, stream>>>(BIG2, vful, BIG1, knn, res);
    gemm_wmma_kernel<0, 64><<<gM, 256, 0, stream>>>(mk(pt_pw, pt_pb, efeat, MPTS, res));
    add_kernel<<<(unsigned)((CM + 255) / 256), 256, 0, stream>>>(efeat, sf, CM);

    // 6. position embedding: fused local-frame input
    gemm_wmma_kernel<3, 4><<<gMK, 256, 0, stream>>>(mk(em_c1w, em_c1b, BIG0, (int)MK));
    do_gn(BIG0, em_gng, em_gnb, MK);
    gemm_wmma_kernel<0, 64><<<gMK, 256, 0, stream>>>(mk(em_c2w, em_c2b, BIG1, (int)MK, BIG0));
    gemm_wmma_kernel<0, 64><<<gMK, 256, 0, stream>>>(mk(at_c1w, at_c1b, BIG2, (int)MK, BIG1));
    do_gn(BIG2, at_gng, at_gnb, MK);
    gemm_wmma_kernel<0, 64><<<gMK, 256, 0, stream>>>(mk(at_c2w, at_c2b, BIG0, (int)MK, BIG2));
    softmax_kernel<<<(unsigned)((CM + 255) / 256), 256, 0, stream>>>(BIG0, maskb, 1.0f);
    wsum_pe_kernel<<<(unsigned)((CM + 255) / 256), 256, 0, stream>>>(BIG1, BIG0, epos);

    // 7. density MLP on downsample_num
    gemm_wmma_kernel<0, 1><<<gM, 256, 0, stream>>>(mk(dn_c1w, dn_c1b, hidm, MPTS, o_dnum));
    do_gn(hidm, dn_gng, dn_gnb, MPTS);
    gemm_wmma_kernel<0, 64><<<gM, 256, 0, stream>>>(mk(dn_c2w, dn_c2b, edens, MPTS, hidm));

    // 8. final: post(concat(e_feat, e_pos, e_dens)) + identity (gathered feats)
    gemm_wmma_kernel<4, 192><<<gM, 256, 0, stream>>>(
        mk(po_w, po_b, o_feat, MPTS, efeat, epos, edens));
    addgather_kernel<<<(unsigned)((CM + 255) / 256), 256, 0, stream>>>(o_feat, feats, sidx);
}
